// DecoderLayer_24764781429447
// MI455X (gfx1250) — compile-verified, matrix operand order checked by reading, not run on hardware
//
#include <hip/hip_runtime.h>
#include <hip/hip_bf16.h>
#include <math.h>

// Problem constants (from reference)
#define N_NODES 4096
#define KNBR    48
#define CDIM    128
#define ECTXD   384
#define DIN     512   // C + ECTX
#define HIDD    512
#define LN_EPS  1e-5f
#define AGG_SCALE 30.0f

typedef _Float16 v16h __attribute__((ext_vector_type(16)));
typedef _Float16 v8h  __attribute__((ext_vector_type(8)));
typedef _Float16 v4h  __attribute__((ext_vector_type(4)));
typedef float    v8f  __attribute__((ext_vector_type(8)));

__device__ __forceinline__ float gelu_exact(float x) {
    // exact gelu: 0.5*x*(1+erf(x/sqrt(2)))
    return 0.5f * x * (1.0f + erff(x * 0.70710678118654752440f));
}

// Load one 16x32 f16 A/B fragment (CDNA5 WMMA layout, ISA 7.12.2):
// per lane: 8 halves at (row, k0 + 8*halfgroup) and 8 halves at +16.
// p0 must be 16B aligned (all our strides are multiples of 8 halves).
__device__ __forceinline__ v16h frag_ab(const _Float16* p0) {
    union { v16h v; v8h h[2]; } u;
    u.h[0] = *(const v8h*)(p0);
    u.h[1] = *(const v8h*)(p0 + 16);
    return u.v;
}

// MT vertically-stacked 16x16 output tiles sharing one B (weight) n-tile:
//   D[mi] = A(m0+16*mi .. +16, KDIM) * B^T(n0..n0+16) + bias
// B fragment is loaded ONCE per k-step and feeds MT WMMAs.
// A row-major [rows x lda] (LDS f16); B row-major weights [out x ldb]
// (global f16), B(k,n) = W[n][k]. bias_col is this lane's column bias.
template <int KDIM, int MT>
__device__ __forceinline__ void gemm_tiles(const _Float16* __restrict__ A, int lda, int m0,
                                           const _Float16* __restrict__ B, int ldb, int n0,
                                           float bias_col, int lane, v8f acc[MT]) {
#pragma unroll
    for (int mi = 0; mi < MT; ++mi)
#pragma unroll
        for (int i = 0; i < 8; ++i) acc[mi][i] = bias_col;
    const int ar = m0 + (lane & 15);
    const int br = n0 + (lane & 15);
    const int hg = (lane >> 4) * 8;
#pragma unroll
    for (int k0 = 0; k0 < KDIM; k0 += 32) {
        v16h b = frag_ab(B + br * ldb + k0 + hg);
#pragma unroll
        for (int mi = 0; mi < MT; ++mi) {
            v16h a = frag_ab(A + (ar + 16 * mi) * lda + k0 + hg);
            acc[mi] = __builtin_amdgcn_wmma_f32_16x16x32_f16(
                /*neg_a=*/false, a, /*neg_b=*/false, b,
                /*c_mod=*/(short)0, acc[mi], /*reuse_a=*/false, /*reuse_b=*/false);
        }
    }
}

// ---------------- kernel 0: convert all weights f32 -> f16 into workspace ----
__global__ __launch_bounds__(256) void prepack_kernel(
    const float* __restrict__ W1, const float* __restrict__ W2,
    const float* __restrict__ W3, const float* __restrict__ Wd1,
    const float* __restrict__ Wd2, _Float16* __restrict__ dst) {
    const int S1 = CDIM * DIN;          // 65536
    const int S2 = S1 + CDIM * CDIM;    // +16384
    const int S3 = S2 + CDIM * CDIM;    // +16384
    const int S4 = S3 + HIDD * CDIM;    // +65536
    const int S5 = S4 + CDIM * HIDD;    // +65536 = 229376
    int i = blockIdx.x * 256 + threadIdx.x;
    if (i < S1)      dst[i] = (_Float16)W1[i];
    else if (i < S2) dst[i] = (_Float16)W2[i - S1];
    else if (i < S3) dst[i] = (_Float16)W3[i - S2];
    else if (i < S4) dst[i] = (_Float16)Wd1[i - S3];
    else if (i < S5) dst[i] = (_Float16)Wd2[i - S4];
}

// ---------------- kernel 1: per-node message MLP + masked aggregation --------
// grid = N nodes, block = 256 threads (8 wave32). One node per block.
__global__ __launch_bounds__(256) void msg_kernel(
    const float* __restrict__ nf,     // (N, 128)
    const float* __restrict__ lef,    // (N, 48, 384)
    const float* __restrict__ amask,  // (N, 48)
    const _Float16* __restrict__ w1h, const float* __restrict__ b1,
    const _Float16* __restrict__ w2h, const float* __restrict__ b2,
    const _Float16* __restrict__ w3h, const float* __restrict__ b3,
    float* __restrict__ x1)           // (N, 128) pre-LN1 residual sum
{
    constexpr int ALD = 520;  // 512 + 8-half pad, keeps 16B alignment
    constexpr int HLD = 136;  // 128 + 8-half pad
    __shared__ __attribute__((aligned(16))) _Float16 act[KNBR * ALD]; // 49920 B
    __shared__ __attribute__((aligned(16))) _Float16 h1[KNBR * HLD]; // 13056 B
    __shared__ __attribute__((aligned(16))) _Float16 h2[KNBR * HLD]; // 13056 B

    const int tid = threadIdx.x;
    const int lane = tid & 31;
    const int wave = tid >> 5;
    const int node = blockIdx.x;

    if (tid == 0 && node + 1 < N_NODES) {
        // warm next node's edge block toward GL2 (global_prefetch_b8)
        __builtin_prefetch(lef + (size_t)(node + 1) * KNBR * ECTXD, 0, 1);
    }

    // --- stage activations: [48 rows x 512] = [node_feat | edge_ctx] as f16 ---
    // node features: read float4, pack 4 halves, one ds_store_b64
    const float* nrow = nf + (size_t)node * CDIM;
    const float4* nrow4 = (const float4*)nrow;
    for (int i = tid; i < KNBR * (CDIM / 4); i += 256) {
        int k = i >> 5, c4 = i & 31;
        float4 v = nrow4[c4];
        v4h h; h.x = (_Float16)v.x; h.y = (_Float16)v.y;
        h.z = (_Float16)v.z; h.w = (_Float16)v.w;
        *(v4h*)&act[k * ALD + c4 * 4] = h;
    }
    const float4* ef4 = (const float4*)(lef + (size_t)node * KNBR * ECTXD);
    for (int i = tid; i < KNBR * (ECTXD / 4); i += 256) {
        int k = i / (ECTXD / 4), c4 = i % (ECTXD / 4);
        float4 v = ef4[i];
        v4h h; h.x = (_Float16)v.x; h.y = (_Float16)v.y;
        h.z = (_Float16)v.z; h.w = (_Float16)v.w;
        *(v4h*)&act[k * ALD + CDIM + c4 * 4] = h;
    }
    __syncthreads();

    // --- layer 1: gelu((48x512) @ W1^T + b1) -> h1 (48x128). wave owns 16 cols,
    // 3 M-tiles computed together against one shared B fragment stream.
    {
        const int n0 = wave * 16;
        const int col = n0 + (lane & 15);
        v8f acc[3];
        gemm_tiles<DIN, 3>(act, ALD, 0, w1h, DIN, n0, b1[col], lane, acc);
#pragma unroll
        for (int mt = 0; mt < 3; ++mt)
#pragma unroll
            for (int i = 0; i < 8; ++i) {
                int row = mt * 16 + (lane >> 4) * 8 + i;
                h1[row * HLD + col] = (_Float16)gelu_exact(acc[mt][i]);
            }
    }
    __syncthreads();

    // --- layer 2: gelu((48x128) @ W2^T + b2) -> h2
    {
        const int n0 = wave * 16;
        const int col = n0 + (lane & 15);
        v8f acc[3];
        gemm_tiles<CDIM, 3>(h1, HLD, 0, w2h, CDIM, n0, b2[col], lane, acc);
#pragma unroll
        for (int mt = 0; mt < 3; ++mt)
#pragma unroll
            for (int i = 0; i < 8; ++i) {
                int row = mt * 16 + (lane >> 4) * 8 + i;
                h2[row * HLD + col] = (_Float16)gelu_exact(acc[mt][i]);
            }
    }
    __syncthreads();

    // --- layer 3 + masked column-sum over the 48 edges; wave owns its 16 cols,
    // so the whole reduction stays inside one wave (shfl across lane halves).
    {
        const int n0 = wave * 16;
        const int col = n0 + (lane & 15);
        const float* am = amask + (size_t)node * KNBR;
        v8f acc[3];
        gemm_tiles<CDIM, 3>(h2, HLD, 0, w3h, CDIM, n0, b3[col], lane, acc);
        float colsum = 0.0f;
#pragma unroll
        for (int mt = 0; mt < 3; ++mt)
#pragma unroll
            for (int i = 0; i < 8; ++i) {
                int row = mt * 16 + (lane >> 4) * 8 + i;
                colsum += acc[mt][i] * am[row];
            }
        colsum += __shfl_xor(colsum, 16, 32); // combine lane-half partials
        if (lane < 16) {
            x1[(size_t)node * CDIM + col] = nrow[col] + colsum * (1.0f / AGG_SCALE);
        }
    }
}

// ---------------- kernel 2: LN1 -> dense MLP -> residual -> LN2 -> mask ------
// grid = N/64, block = 256 threads. 64 nodes per block (4 WMMA M-tiles).
__global__ __launch_bounds__(256) void dense_kernel(
    const float* __restrict__ x1, const float* __restrict__ nmask,
    const float* __restrict__ g1, const float* __restrict__ be1,
    const _Float16* __restrict__ wd1h, const float* __restrict__ bd1,
    const _Float16* __restrict__ wd2h, const float* __restrict__ bd2,
    const float* __restrict__ g2, const float* __restrict__ be2,
    float* __restrict__ out)
{
    constexpr int XLD = 132;   // f32 row stride (bank-conflict pad)
    constexpr int XHL = 136;   // f16 row stride (16B aligned)
    constexpr int HLD = 520;   // f16 hidden stride
    __shared__ __attribute__((aligned(16))) float    xf[64 * XLD]; // 33792 B
    __shared__ __attribute__((aligned(16))) _Float16 xh[64 * XHL]; // 17408 B
    __shared__ __attribute__((aligned(16))) _Float16 hh[64 * HLD]; // 66560 B

    const int tid = threadIdx.x;
    const int lane = tid & 31;
    const int wave = tid >> 5;
    const int nb = blockIdx.x * 64;

    // --- LayerNorm 1: 4 lanes per row, combine with shfl_xor(1),(2) ---
    {
        const int row = tid >> 2, q = tid & 3;
        const float* xr = x1 + (size_t)(nb + row) * CDIM;
        float s = 0.0f, ss = 0.0f;
        for (int c = q * 32; c < q * 32 + 32; ++c) { float v = xr[c]; s += v; ss += v * v; }
        s += __shfl_xor(s, 1, 32); ss += __shfl_xor(ss, 1, 32);
        s += __shfl_xor(s, 2, 32); ss += __shfl_xor(ss, 2, 32);
        float mu  = s * (1.0f / 128.0f);
        float inv = rsqrtf(ss * (1.0f / 128.0f) - mu * mu + LN_EPS);
        for (int c = q * 32; c < q * 32 + 32; ++c) {
            float v = (xr[c] - mu) * inv * g1[c] + be1[c];
            xf[row * XLD + c] = v;
            xh[row * XHL + c] = (_Float16)v;
        }
    }
    __syncthreads();

    // --- d1: gelu((64x128) @ Wd1^T + bd1) -> hh (64x512)
    // wave owns n-tiles {wave, wave+8, wave+16, wave+24}; 4 M-tiles share B.
    for (int nt = wave; nt < 32; nt += 8) {
        const int n0 = nt * 16;
        const int col = n0 + (lane & 15);
        v8f acc[4];
        gemm_tiles<CDIM, 4>(xh, XHL, 0, wd1h, CDIM, n0, bd1[col], lane, acc);
#pragma unroll
        for (int mt = 0; mt < 4; ++mt)
#pragma unroll
            for (int i = 0; i < 8; ++i) {
                int row = mt * 16 + (lane >> 4) * 8 + i;
                hh[row * HLD + col] = (_Float16)gelu_exact(acc[mt][i]);
            }
    }
    __syncthreads();

    // --- d2: (64x512) @ Wd2^T + bd2, residual add into xf (each element owned
    // by exactly one lane slot, so in-place is race-free). wave owns n-tile wave.
    {
        const int n0 = wave * 16;
        const int col = n0 + (lane & 15);
        v8f acc[4];
        gemm_tiles<HIDD, 4>(hh, HLD, 0, wd2h, HIDD, n0, bd2[col], lane, acc);
#pragma unroll
        for (int mt = 0; mt < 4; ++mt)
#pragma unroll
            for (int i = 0; i < 8; ++i) {
                int row = mt * 16 + (lane >> 4) * 8 + i;
                xf[row * XLD + col] += acc[mt][i];
            }
    }
    __syncthreads();

    // --- LayerNorm 2 + output mask ---
    {
        const int row = tid >> 2, q = tid & 3;
        float s = 0.0f, ss = 0.0f;
        for (int c = q * 32; c < q * 32 + 32; ++c) { float v = xf[row * XLD + c]; s += v; ss += v * v; }
        s += __shfl_xor(s, 1, 32); ss += __shfl_xor(ss, 1, 32);
        s += __shfl_xor(s, 2, 32); ss += __shfl_xor(ss, 2, 32);
        float mu  = s * (1.0f / 128.0f);
        float inv = rsqrtf(ss * (1.0f / 128.0f) - mu * mu + LN_EPS);
        float m = nmask[nb + row];
        float* orow = out + (size_t)(nb + row) * CDIM;
        for (int c = q * 32; c < q * 32 + 32; ++c) {
            float v = (xf[row * XLD + c] - mu) * inv * g2[c] + be2[c];
            orow[c] = m * v;
        }
    }
}

extern "C" void kernel_launch(void* const* d_in, const int* in_sizes, int n_in,
                              void* d_out, int out_size, void* d_ws, size_t ws_size,
                              hipStream_t stream) {
    const float* nf  = (const float*)d_in[0];
    const float* lef = (const float*)d_in[1];
    const float* msk = (const float*)d_in[2];
    const float* am  = (const float*)d_in[3];
    const float* Wm1 = (const float*)d_in[4];
    const float* bm1 = (const float*)d_in[5];
    const float* Wm2 = (const float*)d_in[6];
    const float* bm2 = (const float*)d_in[7];
    const float* Wm3 = (const float*)d_in[8];
    const float* bm3 = (const float*)d_in[9];
    const float* g1  = (const float*)d_in[10];
    const float* be1 = (const float*)d_in[11];
    const float* Wd1 = (const float*)d_in[12];
    const float* bd1 = (const float*)d_in[13];
    const float* Wd2 = (const float*)d_in[14];
    const float* bd2 = (const float*)d_in[15];
    const float* g2  = (const float*)d_in[16];
    const float* be2 = (const float*)d_in[17];
    float* out = (float*)d_out;

    // workspace layout: [f16 weights: 229376 halves][f32 x1: N*C]
    _Float16* wh   = (_Float16*)d_ws;
    _Float16* w1h  = wh;                                    // 128x512
    _Float16* w2h  = w1h + CDIM * DIN;                      // 128x128
    _Float16* w3h  = w2h + CDIM * CDIM;                     // 128x128
    _Float16* wd1h = w3h + CDIM * CDIM;                     // 512x128
    _Float16* wd2h = wd1h + HIDD * CDIM;                    // 128x512
    const size_t W_HALVES = (size_t)CDIM * DIN + 2u * CDIM * CDIM +
                            (size_t)HIDD * CDIM + (size_t)CDIM * HIDD; // 229376
    float* x1 = (float*)((char*)d_ws + W_HALVES * sizeof(_Float16));

    prepack_kernel<<<(int)((W_HALVES + 255) / 256), 256, 0, stream>>>(
        Wm1, Wm2, Wm3, Wd1, Wd2, wh);
    msg_kernel<<<N_NODES, 256, 0, stream>>>(
        nf, lef, am, w1h, bm1, w2h, bm2, w3h, bm3, x1);
    dense_kernel<<<N_NODES / 64, 256, 0, stream>>>(
        x1, msk, g1, be1, wd1h, bd1, wd2h, bd2, g2, be2, out);
}
